// COCOSpeaker_32435593019823
// MI455X (gfx1250) — matrix operand order repaired and checked by reading, not compiled, not run on hardware
//
#include <hip/hip_runtime.h>
#include <hip/hip_bf16.h>

typedef __attribute__((ext_vector_type(16))) _Float16 v16h;
typedef __attribute__((ext_vector_type(8)))  float    v8f;
typedef __attribute__((ext_vector_type(4)))  unsigned int v4u;
typedef __attribute__((ext_vector_type(8)))  int      v8i_;
typedef __attribute__((ext_vector_type(4)))  int      v4i_;

#define BB 1024
#define TT 20
#define VV 10000
#define DD 512
#define HH 64

// ---------------------------------------------------------------------------
// Tensor Data Mover: async 2D tile (rows x rowlen f16 elems, row stride
// `stride_elems`) from global -> LDS at byte offset lds_byte. ISA ch.8 D#.
// ---------------------------------------------------------------------------
__device__ inline void tdm_load_2d(const void* gaddr, unsigned lds_byte,
                                   int rows, int rowlen_elems, int stride_elems) {
  unsigned long long ga = (unsigned long long)(uintptr_t)gaddr;
  v4u g0;
  g0[0] = 1u;                                           // count=1, user desc
  g0[1] = lds_byte;                                     // lds_addr
  g0[2] = (unsigned)(ga & 0xFFFFFFFFull);               // global_addr lo
  g0[3] = (unsigned)((ga >> 32) & 0x1FFFFFFull) | (2u << 30);  // addr hi | type=2
  v8i_ g1;
  g1[0] = (int)(1u << 16);                              // data_size=1 (2 bytes)
  g1[1] = (int)0xFFFF0000u;                             // tensor_dim0 = 0x7fffffff (lo16)
  g1[2] = (int)0xFFFF7FFFu;                             // dim0 hi16 | dim1 lo16
  g1[3] = (int)(0x7FFFu | ((unsigned)rowlen_elems << 16)); // dim1 hi16 | tile_dim0
  g1[4] = rows;                                         // tile_dim1 | tile_dim2=0
  g1[5] = stride_elems;                                 // tensor_dim0_stride lo32
  g1[6] = 0;                                            // stride0 hi | stride1 lo
  g1[7] = 0;                                            // stride1 hi
  v4i_ z = {0, 0, 0, 0};
#if __clang_major__ >= 23
  v8i_ z8 = {0, 0, 0, 0, 0, 0, 0, 0};
  __builtin_amdgcn_tensor_load_to_lds(g0, g1, z, z, z8, 0);
#else
  __builtin_amdgcn_tensor_load_to_lds(g0, g1, z, z, 0);
#endif
}

// ---------------------------------------------------------------------------
// f32 -> f16 transposed weight conversion: src [K,N] row-major -> dst [N,K]
// ---------------------------------------------------------------------------
__global__ void cvt_t_kernel(const float* __restrict__ src, _Float16* __restrict__ dst,
                             int K, int N) {
  int i = blockIdx.x * blockDim.x + threadIdx.x;
  if (i >= K * N) return;
  int k = i / N, n = i % N;
  dst[(size_t)n * K + k] = (_Float16)src[i];
}

// f32 -> f16 row copy with dest stride/offset (for building concat input)
__global__ void cvt_rows_kernel(const float* __restrict__ src, _Float16* __restrict__ dst,
                                int Mrows, int Ccols, int ldd, int doff) {
  int i = blockIdx.x * blockDim.x + threadIdx.x;
  if (i >= Mrows * Ccols) return;
  int m = i / Ccols, c = i % Ccols;
  dst[(size_t)m * ldd + doff + c] = (_Float16)src[i];
}

__global__ void fill_zero_kernel(unsigned int* p, int nWords) {
  int i = blockIdx.x * blockDim.x + threadIdx.x;
  if (i < nWords) p[i] = 0u;
}

__global__ void actions_to_float_kernel(const int* __restrict__ a, float* __restrict__ out, int n) {
  int i = blockIdx.x * blockDim.x + threadIdx.x;
  if (i < n) out[i] = (float)a[i];
}

// ---------------------------------------------------------------------------
// WMMA GEMM with TDM->LDS staging: D = act(A[M,K]f16 @ Bt[N,K]^T f16 + bias + resid)
// block = 256 threads = 8 waves; wave computes 32x16; block macro-tile 64x64.
// K staged through LDS in chunks of KC=min(128,K) via tensor_load_to_lds.
// grid = (N/64, M/64); requires M%64==0, N%64==0, K%32==0.
// ---------------------------------------------------------------------------
__global__ __launch_bounds__(256) void wmma_gemm_kernel(
    const _Float16* __restrict__ A, const _Float16* __restrict__ Bt,
    const float* __restrict__ bias, const float* __restrict__ resid,
    float* __restrict__ C, _Float16* __restrict__ C16,
    int M, int N, int K, int act) {
  __shared__ _Float16 sA[64 * 128];
  __shared__ _Float16 sB[64 * 128];
  const int KC = (K < 128) ? K : 128;

  int wave = threadIdx.x >> 5;
  int lane = threadIdx.x & 31;
  int wm = wave >> 2;   // 0..1 -> 32 rows each
  int wn = wave & 3;    // 0..3 -> 16 cols each
  int rowBlk = blockIdx.y * 64;
  int colBlk = blockIdx.x * 64;

  int mA  = lane & 15;
  int kbA = (lane >> 4) << 3;   // A frag K-halves: {0-7,16-23} / {8-15,24-31}
  int nB  = lane & 15;
  int kbB = (lane >> 4) << 3;   // B frag K-halves: {0-7,16-23} / {8-15,24-31}

  const _Float16* sArow0 = sA + (size_t)(wm * 32 + mA) * KC + kbA;
  const _Float16* sArow1 = sArow0 + (size_t)16 * KC;
  const _Float16* sBrow  = sB + (size_t)(wn * 16 + nB) * KC + kbB;

  unsigned ldsA = (unsigned)(uintptr_t)(void*)sA;  // LDS byte offset (low 32 bits)
  unsigned ldsB = (unsigned)(uintptr_t)(void*)sB;

  v8f acc0 = {}, acc1 = {};
  for (int kc = 0; kc < K; kc += KC) {
    if (wave == 0) {
      tdm_load_2d(A  + (size_t)rowBlk * K + kc, ldsA, 64, KC, K);
      tdm_load_2d(Bt + (size_t)colBlk * K + kc, ldsB, 64, KC, K);
      __builtin_amdgcn_s_wait_tensorcnt(0);
    }
    __syncthreads();
    for (int k0 = 0; k0 < KC; k0 += 32) {
      v16h a0, a1, b;
#pragma unroll
      for (int i = 0; i < 8; i++) { b[i] = sBrow[k0 + i]; b[8 + i] = sBrow[k0 + 16 + i]; }
#pragma unroll
      for (int i = 0; i < 8; i++) { a0[i] = sArow0[k0 + i]; a0[8 + i] = sArow0[k0 + 16 + i]; }
#pragma unroll
      for (int i = 0; i < 8; i++) { a1[i] = sArow1[k0 + i]; a1[8 + i] = sArow1[k0 + 16 + i]; }
      acc0 = __builtin_amdgcn_wmma_f32_16x16x32_f16(false, a0, false, b, (short)0, acc0, false, false);
      acc1 = __builtin_amdgcn_wmma_f32_16x16x32_f16(false, a1, false, b, (short)0, acc1, false, false);
    }
    __syncthreads();
  }

  int col   = colBlk + wn * 16 + (lane & 15);
  int rbase = rowBlk + wm * 32 + ((lane >> 4) << 3);
  float bi = bias ? bias[col] : 0.0f;
#pragma unroll
  for (int r = 0; r < 8; r++) {
#pragma unroll
    for (int half = 0; half < 2; half++) {
      int row = rbase + r + half * 16;
      float v = (half ? acc1[r] : acc0[r]) + bi;
      if (resid) v += resid[(size_t)row * N + col];
      if (act == 1)      v = 0.5f * v * (1.0f + erff(v * 0.70710678118654752f)); // exact GELU
      else if (act == 2) v = tanhf(v);
      if (C)   C[(size_t)row * N + col]   = v;
      if (C16) C16[(size_t)row * N + col] = (_Float16)v;
    }
  }
}

// ---------------------------------------------------------------------------
// LayerNorm over rows of 512: out16 = LN(x (+ resid)) * g + b
// ---------------------------------------------------------------------------
__global__ __launch_bounds__(256) void ln_kernel(
    const float* __restrict__ x, const float* __restrict__ resid,
    const float* __restrict__ g, const float* __restrict__ bb,
    _Float16* __restrict__ out16) {
  int row = blockIdx.x;
  int tid = threadIdx.x;
  const float* xr = x + (size_t)row * 512;
  float v0 = xr[tid]       + (resid ? resid[(size_t)row * 512 + tid]       : 0.0f);
  float v1 = xr[tid + 256] + (resid ? resid[(size_t)row * 512 + tid + 256] : 0.0f);
  __shared__ float red[256];
  red[tid] = v0 + v1;
  __syncthreads();
  for (int s = 128; s > 0; s >>= 1) { if (tid < s) red[tid] += red[tid + s]; __syncthreads(); }
  float mean = red[0] * (1.0f / 512.0f);
  __syncthreads();
  float d0 = v0 - mean, d1 = v1 - mean;
  red[tid] = d0 * d0 + d1 * d1;
  __syncthreads();
  for (int s = 128; s > 0; s >>= 1) { if (tid < s) red[tid] += red[tid + s]; __syncthreads(); }
  float rs = rsqrtf(red[0] * (1.0f / 512.0f) + 1e-5f);
  float y0 = d0 * rs, y1 = d1 * rs;
  if (g) { y0 = y0 * g[tid] + bb[tid]; y1 = y1 * g[tid + 256] + bb[tid + 256]; }
  out16[(size_t)row * 512 + tid]       = (_Float16)y0;
  out16[(size_t)row * 512 + tid + 256] = (_Float16)y1;
}

// ---------------------------------------------------------------------------
// GRU gates: h = (1-z)*n + z*ix  -> hx16
// ---------------------------------------------------------------------------
__global__ void gru_gate_kernel(const float* __restrict__ gi, const float* __restrict__ gh,
                                const float* __restrict__ ix, _Float16* __restrict__ hx16) {
  int i = blockIdx.x * blockDim.x + threadIdx.x;
  if (i >= BB * DD) return;
  int b = i >> 9, j = i & 511;
  size_t base = (size_t)b * 1536;
  float r = 1.0f / (1.0f + __expf(-(gi[base + j]       + gh[base + j])));
  float z = 1.0f / (1.0f + __expf(-(gi[base + 512 + j] + gh[base + 512 + j])));
  float n = tanhf(gi[base + 1024 + j] + r * gh[base + 1024 + j]);
  float h = (1.0f - z) * n + z * ix[i];
  hx16[i] = (_Float16)h;
}

// embedding gather for next step input: ix = emb[actions[:,t]]
__global__ void gather_emb_kernel(const float* __restrict__ emb, const int* __restrict__ actions,
                                  int t, float* __restrict__ ix32, _Float16* __restrict__ ix16) {
  int i = blockIdx.x * blockDim.x + threadIdx.x;
  if (i >= BB * DD) return;
  int b = i >> 9, j = i & 511;
  int a = actions[b * TT + t];
  float v = emb[(size_t)a * DD + j];
  ix32[i] = v;
  ix16[i] = (_Float16)v;
}

// critic final: values[:,t] = c2[B,64] @ w3[64] + b3
__global__ void critic_val_kernel(const float* __restrict__ c2, const float* __restrict__ w3,
                                  const float* __restrict__ b3, float* __restrict__ out_val,
                                  int t) {
  int row = blockIdx.x * blockDim.x + threadIdx.x;
  if (row >= BB) return;
  float s = 0.0f;
#pragma unroll
  for (int j = 0; j < 64; j++) s += c2[(size_t)row * 64 + j] * w3[j];
  out_val[row * TT + t] = s + b3[0];
}

// ---------------------------------------------------------------------------
// Fused logits GEMM [16,64]x[64,V] + online log-sum-exp per row:
//   lp = logit[action] - logZ ; ent = logZ - E[logit]
// ---------------------------------------------------------------------------
__global__ __launch_bounds__(256) void logits_lse_kernel(
    const _Float16* __restrict__ A, const _Float16* __restrict__ W3t,
    const float* __restrict__ b3, const float* __restrict__ mask,
    const int* __restrict__ actions, float* __restrict__ out_lp,
    float* __restrict__ out_ent, int t) {
  const int NT = VV >> 4;  // 625
  int wave = threadIdx.x >> 5;
  int lane = threadIdx.x & 31;
  int rb = blockIdx.x * 16;
  int mA  = lane & 15;
  int hi8 = (lane >> 4) << 3;

  const _Float16* Ap = A + (size_t)(rb + mA) * 64 + hi8;
  v16h a0, a1;
#pragma unroll
  for (int i = 0; i < 8; i++) {
    a0[i] = Ap[i];      a0[8 + i] = Ap[16 + i];
    a1[i] = Ap[32 + i]; a1[8 + i] = Ap[48 + i];
  }
  int arow[8];
#pragma unroll
  for (int r = 0; r < 8; r++) arow[r] = actions[(rb + r + hi8) * TT + t];

  float m[8], s0[8], s1[8], la[8];
#pragma unroll
  for (int r = 0; r < 8; r++) { m[r] = -INFINITY; s0[r] = 0.f; s1[r] = 0.f; la[r] = -INFINITY; }

  int nB  = lane & 15;
  int kbB = (lane >> 4) << 3;
  for (int nt = wave; nt < NT; nt += 8) {
    int col0 = nt << 4;
    const _Float16* Bp = W3t + (size_t)(col0 + nB) * 64 + kbB;
    v16h b0, b1;
#pragma unroll
    for (int i = 0; i < 8; i++) { b0[i] = Bp[i]; b0[8 + i] = Bp[16 + i]; b1[i] = Bp[32 + i]; b1[8 + i] = Bp[48 + i]; }
    v8f acc = {};
    acc = __builtin_amdgcn_wmma_f32_16x16x32_f16(false, a0, false, b0, (short)0, acc, false, false);
    acc = __builtin_amdgcn_wmma_f32_16x16x32_f16(false, a1, false, b1, (short)0, acc, false, false);
    int col = col0 + nB;
    float bm = b3[col] + mask[col];
#pragma unroll
    for (int r = 0; r < 8; r++) {
      float lv = acc[r] + bm;
      float mn = fmaxf(m[r], lv);
      float sc = __expf(m[r] - mn);
      float e  = __expf(lv - mn);
      s0[r] = s0[r] * sc + e;
      s1[r] = s1[r] * sc + e * lv;
      m[r]  = mn;
      if (col == arow[r]) la[r] = lv;
    }
  }
#pragma unroll
  for (int xm = 1; xm < 16; xm <<= 1) {
#pragma unroll
    for (int r = 0; r < 8; r++) {
      float m2  = __shfl_xor(m[r],  xm, 32);
      float s02 = __shfl_xor(s0[r], xm, 32);
      float s12 = __shfl_xor(s1[r], xm, 32);
      float la2 = __shfl_xor(la[r], xm, 32);
      float mn = fmaxf(m[r], m2);
      float ca = __expf(m[r] - mn), cb = __expf(m2 - mn);
      s0[r] = s0[r] * ca + s02 * cb;
      s1[r] = s1[r] * ca + s12 * cb;
      m[r]  = mn;
      la[r] = fmaxf(la[r], la2);
    }
  }
  __shared__ float sm_m[8][16], sm_s0[8][16], sm_s1[8][16], sm_la[8][16];
  if ((lane & 15) == 0) {
#pragma unroll
    for (int r = 0; r < 8; r++) {
      int rl = r + hi8;
      sm_m[wave][rl] = m[r]; sm_s0[wave][rl] = s0[r];
      sm_s1[wave][rl] = s1[r]; sm_la[wave][rl] = la[r];
    }
  }
  __syncthreads();
  if (threadIdx.x < 16) {
    int rl = threadIdx.x;
    float M0 = sm_m[0][rl], S0 = sm_s0[0][rl], S1 = sm_s1[0][rl], LA = sm_la[0][rl];
    for (int w = 1; w < 8; w++) {
      float m2 = sm_m[w][rl];
      float mn = fmaxf(M0, m2);
      float ca = __expf(M0 - mn), cb = __expf(m2 - mn);
      S0 = S0 * ca + sm_s0[w][rl] * cb;
      S1 = S1 * ca + sm_s1[w][rl] * cb;
      M0 = mn;
      LA = fmaxf(LA, sm_la[w][rl]);
    }
    float logZ = M0 + __logf(S0);
    int row = rb + rl;
    out_lp[row * TT + t]  = LA - logZ;
    out_ent[row * TT + t] = logZ - S1 / S0;
  }
}

// ---------------------------------------------------------------------------
extern "C" void kernel_launch(void* const* d_in, const int* in_sizes, int n_in,
                              void* d_out, int out_size, void* d_ws, size_t ws_size,
                              hipStream_t stream) {
  const float* image   = (const float*)d_in[0];
  const float* boxf    = (const float*)d_in[1];
  const int*   actions = (const int*)d_in[2];
  const float* fr_W  = (const float*)d_in[3];  const float* fr_b  = (const float*)d_in[4];
  const float* b1W1  = (const float*)d_in[5];  const float* b1b1  = (const float*)d_in[6];
  const float* b1W2  = (const float*)d_in[7];  const float* b1b2  = (const float*)d_in[8];
  const float* b2W1  = (const float*)d_in[9];  const float* b2b1  = (const float*)d_in[10];
  const float* b2W2  = (const float*)d_in[11]; const float* b2b2  = (const float*)d_in[12];
  const float* ln_g  = (const float*)d_in[13]; const float* ln_b  = (const float*)d_in[14];
  const float* encW  = (const float*)d_in[15]; const float* encB  = (const float*)d_in[16];
  const float* emb   = (const float*)d_in[17];
  const float* gWih  = (const float*)d_in[18]; const float* gbih  = (const float*)d_in[19];
  const float* gWhh  = (const float*)d_in[20]; const float* gbhh  = (const float*)d_in[21];
  const float* aW1   = (const float*)d_in[22]; const float* ab1   = (const float*)d_in[23];
  const float* aW2   = (const float*)d_in[24]; const float* ab2   = (const float*)d_in[25];
  const float* aW3   = (const float*)d_in[26]; const float* ab3   = (const float*)d_in[27];
  const float* cW1   = (const float*)d_in[28]; const float* cb1   = (const float*)d_in[29];
  const float* cW2   = (const float*)d_in[30]; const float* cb2   = (const float*)d_in[31];
  const float* cW3   = (const float*)d_in[32]; const float* cb3   = (const float*)d_in[33];
  const float* wmask = (const float*)d_in[34];

  float* out_actions = (float*)d_out;
  float* out_lp      = out_actions + BB * TT;
  float* out_ent     = out_lp + BB * TT;
  float* out_val     = out_ent + BB * TT;

  // ---- workspace bump allocator ----
  char* ws = (char*)d_ws;
  size_t off = 0;
  auto alloc = [&](size_t bytes) -> void* {
    void* p = ws + off;
    off = (off + bytes + 255) & ~(size_t)255;
    return p;
  };
  _Float16* wt_fr  = (_Float16*)alloc((size_t)2304 * 512 * 2);
  _Float16* wt_11  = (_Float16*)alloc((size_t)512 * 512 * 2);
  _Float16* wt_12  = (_Float16*)alloc((size_t)512 * 512 * 2);
  _Float16* wt_21  = (_Float16*)alloc((size_t)512 * 512 * 2);
  _Float16* wt_22  = (_Float16*)alloc((size_t)512 * 512 * 2);
  _Float16* wt_enc = (_Float16*)alloc((size_t)512 * 512 * 2);
  _Float16* wt_ih  = (_Float16*)alloc((size_t)512 * 1536 * 2);
  _Float16* wt_hh  = (_Float16*)alloc((size_t)512 * 1536 * 2);
  _Float16* wt_a1  = (_Float16*)alloc((size_t)512 * 64 * 2);
  _Float16* wt_a2  = (_Float16*)alloc((size_t)64 * 64 * 2);
  _Float16* wt_a3  = (_Float16*)alloc((size_t)64 * VV * 2);
  _Float16* wt_c1  = (_Float16*)alloc((size_t)512 * 64 * 2);
  _Float16* wt_c2  = (_Float16*)alloc((size_t)64 * 64 * 2);
  _Float16* xcat16 = (_Float16*)alloc((size_t)BB * 2304 * 2);
  float*    x32    = (float*)   alloc((size_t)BB * DD * 4);
  _Float16* y16    = (_Float16*)alloc((size_t)BB * DD * 2);
  _Float16* t16    = (_Float16*)alloc((size_t)BB * DD * 2);
  float*    u32    = (float*)   alloc((size_t)BB * DD * 4);
  _Float16* y2_16  = (_Float16*)alloc((size_t)BB * DD * 2);
  _Float16* t2_16  = (_Float16*)alloc((size_t)BB * DD * 2);
  float*    blk32  = (float*)   alloc((size_t)BB * DD * 4);
  _Float16* ln3_16 = (_Float16*)alloc((size_t)BB * DD * 2);
  _Float16* hx16   = (_Float16*)alloc((size_t)BB * DD * 2);
  float*    ix32   = (float*)   alloc((size_t)BB * DD * 4);
  _Float16* ix16   = (_Float16*)alloc((size_t)BB * DD * 2);
  float*    gi32   = (float*)   alloc((size_t)BB * 1536 * 4);
  float*    gh32   = (float*)   alloc((size_t)BB * 1536 * 4);
  _Float16* a1_16  = (_Float16*)alloc((size_t)BB * HH * 2);
  _Float16* a2_16  = (_Float16*)alloc((size_t)BB * HH * 2);
  _Float16* c1_16  = (_Float16*)alloc((size_t)BB * HH * 2);
  float*    c2_32  = (float*)   alloc((size_t)BB * HH * 4);
  (void)ws_size; (void)n_in; (void)in_sizes; (void)out_size;

  auto cvt = [&](const float* src, _Float16* dst, int K, int N) {
    int n = K * N;
    cvt_t_kernel<<<(n + 255) / 256, 256, 0, stream>>>(src, dst, K, N);
  };
  auto gemm = [&](const _Float16* A, const _Float16* Bt, const float* bias,
                  const float* resid, float* C, _Float16* C16,
                  int M, int N, int K, int act) {
    dim3 g(N / 64, M / 64);
    wmma_gemm_kernel<<<g, 256, 0, stream>>>(A, Bt, bias, resid, C, C16, M, N, K, act);
  };

  cvt(fr_W, wt_fr, 2304, 512);
  cvt(b1W1, wt_11, 512, 512);  cvt(b1W2, wt_12, 512, 512);
  cvt(b2W1, wt_21, 512, 512);  cvt(b2W2, wt_22, 512, 512);
  cvt(encW, wt_enc, 512, 512);
  cvt(gWih, wt_ih, 512, 1536); cvt(gWhh, wt_hh, 512, 1536);
  cvt(aW1, wt_a1, 512, 64);    cvt(aW2, wt_a2, 64, 64);  cvt(aW3, wt_a3, 64, VV);
  cvt(cW1, wt_c1, 512, 64);    cvt(cW2, wt_c2, 64, 64);

  {
    int n = BB * 2048;
    cvt_rows_kernel<<<(n + 255) / 256, 256, 0, stream>>>(image, xcat16, BB, 2048, 2304, 0);
    n = BB * 256;
    cvt_rows_kernel<<<(n + 255) / 256, 256, 0, stream>>>(boxf, xcat16, BB, 256, 2304, 2048);
  }
  actions_to_float_kernel<<<(BB * TT + 255) / 256, 256, 0, stream>>>(actions, out_actions, BB * TT);

  // ---- encoder ----
  gemm(xcat16, wt_fr, fr_b, nullptr, x32, nullptr, BB, 512, 2304, 0);       // x
  ln_kernel<<<BB, 256, 0, stream>>>(x32, nullptr, nullptr, nullptr, y16);   // y = LN(x)
  gemm(y16, wt_11, b1b1, nullptr, nullptr, t16, BB, 512, 512, 1);           // gelu(y@W1+b1)
  gemm(t16, wt_12, b1b2, nullptr, u32, nullptr, BB, 512, 512, 0);           // @W2+b2
  ln_kernel<<<BB, 256, 0, stream>>>(u32, x32, nullptr, nullptr, y2_16);     // LN(u + x)
  gemm(y2_16, wt_21, b2b1, nullptr, nullptr, t2_16, BB, 512, 512, 1);
  gemm(t2_16, wt_22, b2b2, x32, blk32, nullptr, BB, 512, 512, 0);           // blk = x + ...
  ln_kernel<<<BB, 256, 0, stream>>>(blk32, nullptr, ln_g, ln_b, ln3_16);    // LN(blk)*g+b
  gemm(ln3_16, wt_enc, encB, nullptr, nullptr, hx16, BB, 512, 512, 0);      // img_rep -> hx

  fill_zero_kernel<<<(BB * DD + 255) / 256, 256, 0, stream>>>((unsigned int*)ix32, BB * DD);
  fill_zero_kernel<<<(BB * DD / 2 + 255) / 256, 256, 0, stream>>>((unsigned int*)ix16, BB * DD / 2);

  // ---- GRU loop ----
  for (int t = 0; t < TT; t++) {
    gemm(hx16, wt_ih, gbih, nullptr, gi32, nullptr, BB, 1536, 512, 0);      // gi = hx@Wih
    gemm(ix16, wt_hh, gbhh, nullptr, gh32, nullptr, BB, 1536, 512, 0);      // gh = ix@Whh
    gru_gate_kernel<<<(BB * DD + 255) / 256, 256, 0, stream>>>(gi32, gh32, ix32, hx16);
    // actor
    gemm(hx16, wt_a1, ab1, nullptr, nullptr, a1_16, BB, 64, 512, 2);
    gemm(a1_16, wt_a2, ab2, nullptr, nullptr, a2_16, BB, 64, 64, 2);
    logits_lse_kernel<<<BB / 16, 256, 0, stream>>>(a2_16, wt_a3, ab3, wmask, actions,
                                                   out_lp, out_ent, t);
    // critic
    gemm(hx16, wt_c1, cb1, nullptr, nullptr, c1_16, BB, 64, 512, 2);
    gemm(c1_16, wt_c2, cb2, nullptr, c2_32, nullptr, BB, 64, 64, 2);
    critic_val_kernel<<<(BB + 255) / 256, 256, 0, stream>>>(c2_32, cW3, cb3, out_val, t);
    // next input embedding
    gather_emb_kernel<<<(BB * DD + 255) / 256, 256, 0, stream>>>(emb, actions, t, ix32, ix16);
  }
}
